// mLSTMBlock_6030134084211
// MI455X (gfx1250) — matrix operand order repaired and checked by reading, not compiled, not run on hardware
//
#include <hip/hip_runtime.h>
#include <hip/hip_bf16.h>

// ---------------------------------------------------------------------------
// mLSTM block for MI455X (gfx1250): bf16 WMMA GEMMs (async-LDS double-buffered)
// + register-resident sequential scan.
// ---------------------------------------------------------------------------

typedef __attribute__((ext_vector_type(16))) __bf16 v16bf;
typedef __attribute__((ext_vector_type(8)))  float  v8f;

#define TILE_M 128
#define TILE_N 128
#define TILE_K 32
#define LDS_PAD 8   // +8 bf16 => row stride 40 bf16 = 80B: 16B-aligned rows for
                    // async B128 stores, 20-dword stride -> conflict-free ds reads

union FragU { v16bf v; unsigned u[8]; };

// Async copy global -> LDS, 16B per lane, ASYNCcnt-tracked (CDNA5 TDM-lite path)
#define ASYNC_B128(ldsoff, gptr)                                              \
    asm volatile("global_load_async_to_lds_b128 %0, %1, off"                  \
                 :: "v"(ldsoff), "v"(gptr) : "memory")

static __device__ __forceinline__ void wait_async0() {
#if __has_builtin(__builtin_amdgcn_s_wait_asynccnt)
    __builtin_amdgcn_s_wait_asynccnt(0);
#else
    asm volatile("s_wait_asynccnt 0x0" ::: "memory");
#endif
}

// ---------------------------------------------------------------------------
// Generic GEMM:  C[M,N] = A[M,K](bf16, row-major) * B[N,K](bf16, row-major)^T
//                (+ optional fp32 bias[N]), C fp32 row-major.
// 256 threads = 8 waves arranged 4(M) x 2(N); block tile 128x128; K step 32.
// Each wave owns a 32(M) x 64(N) region = 2x4 v_wmma_f32_16x16x32_bf16 tiles.
// LDS tiles are double-buffered and filled with global_load_async_to_lds_b128
// so the next K-tile streams in while the current one feeds the WMMAs.
// ---------------------------------------------------------------------------
__global__ __launch_bounds__(256) void gemm_bf16_wmma(
    const __bf16* __restrict__ A, const __bf16* __restrict__ B,
    const float* __restrict__ bias, float* __restrict__ C,
    int M, int N, int K)
{
    __shared__ __bf16 As[2][TILE_M][TILE_K + LDS_PAD];
    __shared__ __bf16 Bs[2][TILE_N][TILE_K + LDS_PAD];

    const int t      = threadIdx.x;
    const int lane   = t & 31;
    const int wid    = t >> 5;
    const int wm     = wid >> 1;       // 0..3 -> M offset wm*32
    const int wn     = wid & 1;        // 0..1 -> N offset wn*64
    const int laneLo = lane & 15;
    const int laneHi = lane >> 4;      // 0/1

    const int mTile = blockIdx.y * TILE_M;
    const int nTile = blockIdx.x * TILE_N;

    v8f acc[2][4] = {};

    // Per-thread async-copy assignments: 1024 chunks of 16B (512 A + 512 B),
    // 4 chunks/thread.  chunk -> (row = c>>2, quad = c&3): 4 x 16B per 64B row.
    const int c0row = t >> 2,        c0q = t & 3;          // chunk t
    const int c1row = (t + 256) >> 2, c1q = (t + 256) & 3; // chunk t+256

    // ---- prologue: fill buffer 0 for k0 = 0 ----
    {
        ASYNC_B128((unsigned)(size_t)&As[0][c0row][c0q * 8],
                   A + (size_t)(mTile + c0row) * K + c0q * 8);
        ASYNC_B128((unsigned)(size_t)&As[0][c1row][c1q * 8],
                   A + (size_t)(mTile + c1row) * K + c1q * 8);
        ASYNC_B128((unsigned)(size_t)&Bs[0][c0row][c0q * 8],
                   B + (size_t)(nTile + c0row) * K + c0q * 8);
        ASYNC_B128((unsigned)(size_t)&Bs[0][c1row][c1q * 8],
                   B + (size_t)(nTile + c1row) * K + c1q * 8);
    }
    wait_async0();
    __syncthreads();

    for (int k0 = 0; k0 < K; k0 += TILE_K) {
        const int buf = (k0 >> 5) & 1;

        // ---- prefetch next K tile into the other buffer (overlaps WMMA) ----
        if (k0 + TILE_K < K) {
            const int nb = buf ^ 1;
            const int kn = k0 + TILE_K;
            ASYNC_B128((unsigned)(size_t)&As[nb][c0row][c0q * 8],
                       A + (size_t)(mTile + c0row) * K + kn + c0q * 8);
            ASYNC_B128((unsigned)(size_t)&As[nb][c1row][c1q * 8],
                       A + (size_t)(mTile + c1row) * K + kn + c1q * 8);
            ASYNC_B128((unsigned)(size_t)&Bs[nb][c0row][c0q * 8],
                       B + (size_t)(nTile + c0row) * K + kn + c0q * 8);
            ASYNC_B128((unsigned)(size_t)&Bs[nb][c1row][c1q * 8],
                       B + (size_t)(nTile + c1row) * K + kn + c1q * 8);
        }

        // ---- build fragments from LDS per documented 16-bit layouts ----
        // A 16x32: M = lane&15 ; VGPR p<4: K = laneHi*8 + 2p(+1);
        //          p>=4: K = 16 + laneHi*8 + 2(p-4)(+1)
        FragU afr[2];
        #pragma unroll
        for (int mt = 0; mt < 2; ++mt) {
            const int m = wm * 32 + mt * 16 + laneLo;
            const unsigned* row = (const unsigned*)&As[buf][m][0];
            const int kb2 = laneHi * 4;
            #pragma unroll
            for (int pd = 0; pd < 4; ++pd) {
                afr[mt].u[pd]     = row[kb2 + pd];
                afr[mt].u[pd + 4] = row[8 + kb2 + pd];
            }
        }
        // B 32x16: N = lane&15 ; element e<8 -> K = laneHi*8 + e; e>=8 -> K = 16 + laneHi*8 + (e-8)
        FragU bfr[4];
        #pragma unroll
        for (int nt = 0; nt < 4; ++nt) {
            const int n = wn * 64 + nt * 16 + laneLo;
            const unsigned* row = (const unsigned*)&Bs[buf][n][0];
            #pragma unroll
            for (int pd = 0; pd < 8; ++pd)
                bfr[nt].u[pd] = row[(pd >> 2) * 8 + laneHi * 4 + (pd & 3)];
        }

        // ---- 8 WMMA ops per K step ----
        #pragma unroll
        for (int mt = 0; mt < 2; ++mt)
            #pragma unroll
            for (int nt = 0; nt < 4; ++nt)
                acc[mt][nt] = __builtin_amdgcn_wmma_f32_16x16x32_bf16(
                    false, afr[mt].v, false, bfr[nt].v,
                    (short)0, acc[mt][nt], false, false);

        // next-tile async fill must be complete & all waves past their reads
        wait_async0();
        __syncthreads();
    }

    // ---- epilogue: VGPR r -> lanes0-15 M=r, lanes16-31 M=r+8; N=lane&15 ----
    #pragma unroll
    for (int mt = 0; mt < 2; ++mt) {
        #pragma unroll
        for (int nt = 0; nt < 4; ++nt) {
            const int col = nTile + wn * 64 + nt * 16 + laneLo;
            const float bs = bias ? bias[col] : 0.0f;
            #pragma unroll
            for (int r = 0; r < 8; ++r) {
                const int rowm = mTile + wm * 32 + mt * 16 + laneHi * 8 + r;
                C[(size_t)rowm * N + col] = acc[mt][nt][r] + bs;
            }
        }
    }
}

// ---------------------------------------------------------------------------
// fp32 -> bf16 conversion (weights + activations)
// ---------------------------------------------------------------------------
__global__ void f32_to_bf16(const float* __restrict__ in, __bf16* __restrict__ out, int n)
{
    int i = blockIdx.x * blockDim.x + threadIdx.x;
    if (i < n) out[i] = (__bf16)in[i];
}

// Build concatenated bias for the fused 6-way projection GEMM (N = 6144)
// order: q(0), k(1024), v(2048), ig(3072), fg(4096), og(5120)
__global__ void build_bias6(const float* __restrict__ ib, const float* __restrict__ fb,
                            const float* __restrict__ ob, float* __restrict__ bias6)
{
    int j = blockIdx.x * blockDim.x + threadIdx.x;
    if (j >= 6144) return;
    float v = 0.0f;
    if      (j >= 5120) v = ob[j - 5120];
    else if (j >= 4096) v = fb[j - 4096];
    else if (j >= 3072) v = ib[j - 3072];
    bias6[j] = v;
}

// Split in_proj output: x_inner -> bf16 (feeds stage-2 GEMM), x_gate -> sigmoid fp32
__global__ void split_xp(const float* __restrict__ xp, __bf16* __restrict__ x_inner,
                         float* __restrict__ gsig, int n /* rows*1024 */)
{
    int i = blockIdx.x * blockDim.x + threadIdx.x;
    if (i >= n) return;
    int r = i >> 10, c = i & 1023;
    x_inner[i] = (__bf16)xp[(size_t)r * 2048 + c];
    float g = xp[(size_t)r * 2048 + 1024 + c];
    gsig[i] = 1.0f / (1.0f + __expf(-g));
}

// ---------------------------------------------------------------------------
// Activation epilogue: per-head RMSNorm(q,k), exp(i), sigmoid(f,o).
// raw6: (row, 6144) with [q|k|v|ig|fg|og] blocks (bias already added by GEMM).
// q,k,v,ig,fg written in scan layout (b,h,l,d); og kept in row layout.
// ---------------------------------------------------------------------------
__global__ __launch_bounds__(256) void activate_kernel(
    const float* __restrict__ raw6,
    const float* __restrict__ qnw, const float* __restrict__ knw,
    float* __restrict__ q, float* __restrict__ k, float* __restrict__ v,
    float* __restrict__ ig, float* __restrict__ fg, float* __restrict__ og_row)
{
    const int r = blockIdx.x;            // 0..8191  (= b*4096 + l)
    const int b = r >> 12, l = r & 4095;
    const int t = threadIdx.x;
    const int h = t >> 4;                // 0..15
    const int sub = t & 15;              // 16 threads per head

    __shared__ float pq[16][16], pk[16][16], sq[16], sk[16];

    const float* base = raw6 + (size_t)r * 6144;
    float qv[4], kv[4], sumq = 0.0f, sumk = 0.0f;
    #pragma unroll
    for (int j = 0; j < 4; ++j) {
        int d = sub * 4 + j;
        qv[j] = base[h * 64 + d];
        kv[j] = base[1024 + h * 64 + d];
        sumq = fmaf(qv[j], qv[j], sumq);
        sumk = fmaf(kv[j], kv[j], sumk);
    }
    pq[h][sub] = sumq; pk[h][sub] = sumk;
    __syncthreads();
    if (sub == 0) {
        float a = 0.0f, bb = 0.0f;
        #pragma unroll
        for (int i = 0; i < 16; ++i) { a += pq[h][i]; bb += pk[h][i]; }
        sq[h] = rsqrtf(a * (1.0f / 64.0f) + 1e-6f);
        sk[h] = rsqrtf(bb * (1.0f / 64.0f) + 1e-6f);
    }
    __syncthreads();

    const size_t obase = (((size_t)(b * 16 + h)) * 4096 + l) * 64;
    const float qs = sq[h], ks = sk[h];
    #pragma unroll
    for (int j = 0; j < 4; ++j) {
        int d = sub * 4 + j;
        q[obase + d]  = qv[j] * qs * qnw[d];
        k[obase + d]  = kv[j] * ks * knw[d];
        v[obase + d]  = base[2048 + h * 64 + d];
        ig[obase + d] = __expf(base[3072 + h * 64 + d]);
        fg[obase + d] = 1.0f / (1.0f + __expf(-base[4096 + h * 64 + d]));
        og_row[(size_t)r * 1024 + h * 64 + d] = 1.0f / (1.0f + __expf(-base[5120 + h * 64 + d]));
    }
}

// ---------------------------------------------------------------------------
// mLSTM scan: one block per (b,h); 64x64 state C in registers (16 elems/thread).
// thread t: column e = t&63, row group g = t>>6 owns rows d = g*16..g*16+15.
// Fuses * og * sigmoid(x_gate) and bf16 conversion -> hgate feeds final GEMM.
// ---------------------------------------------------------------------------
__global__ __launch_bounds__(256) void mlstm_scan_kernel(
    const float* __restrict__ q, const float* __restrict__ k, const float* __restrict__ v,
    const float* __restrict__ ig, const float* __restrict__ fg,
    const float* __restrict__ og_row, const float* __restrict__ gsig,
    __bf16* __restrict__ hgate)
{
    const int bh = blockIdx.x;           // 0..31
    const int b = bh >> 4, h = bh & 15;
    const int t = threadIdx.x;
    const int e = t & 63, g = t >> 6;
    const size_t base = (size_t)bh * 4096 * 64;

    __shared__ float qs[64], ks[64], vs[64], is[64], fs[64];
    __shared__ float pnum[4][64];
    __shared__ float pden[4];

    float Creg[16], nreg[16];
    #pragma unroll
    for (int j = 0; j < 16; ++j) { Creg[j] = 0.0f; nreg[j] = 0.0f; }

    for (int l = 0; l < 4096; ++l) {
        const size_t gofs = base + (size_t)l * 64;
        if (t < 64)        { qs[t] = q[gofs + t]; fs[t] = fg[gofs + t]; }
        else if (t < 128)  { ks[t - 64]  = k[gofs + t - 64]; }
        else if (t < 192)  { vs[t - 128] = v[gofs + t - 128]; }
        else               { is[t - 192] = ig[gofs + t - 192]; }
        __syncthreads();

        const float ve = vs[e];
        float numpart = 0.0f, denpart = 0.0f;
        #pragma unroll
        for (int j = 0; j < 16; ++j) {
            const int d = g * 16 + j;
            const float ikd = is[d] * ks[d];
            const float c = fmaf(fs[d], Creg[j], ikd * ve);
            Creg[j] = c;
            numpart = fmaf(qs[d], c, numpart);
            const float nn = fmaf(fs[d], nreg[j], ikd);
            nreg[j] = nn;
            denpart = fmaf(qs[d], nn, denpart);
        }
        pnum[g][e] = numpart;
        if (e == 0) pden[g] = denpart;
        __syncthreads();

        if (g == 0) {
            float num = pnum[0][e] + pnum[1][e] + pnum[2][e] + pnum[3][e];
            float den = pden[0] + pden[1] + pden[2] + pden[3];
            den = fmaxf(den, 1.0f);
            const float hv = num / den;
            const size_t ridx = ((size_t)(b * 4096 + l)) * 1024 + h * 64 + e;
            hgate[ridx] = (__bf16)(hv * og_row[ridx] * gsig[ridx]);
        }
        __syncthreads();
    }
}

// ---------------------------------------------------------------------------
// Host-side orchestration
// ---------------------------------------------------------------------------
extern "C" void kernel_launch(void* const* d_in, const int* in_sizes, int n_in,
                              void* d_out, int out_size, void* d_ws, size_t ws_size,
                              hipStream_t stream)
{
    const float* x    = (const float*)d_in[0];
    const float* w_in = (const float*)d_in[1];
    const float* wq   = (const float*)d_in[2];
    const float* wk   = (const float*)d_in[3];
    const float* wv   = (const float*)d_in[4];
    const float* wi   = (const float*)d_in[5];
    const float* bi   = (const float*)d_in[6];
    const float* wf   = (const float*)d_in[7];
    const float* bfb  = (const float*)d_in[8];
    const float* wo   = (const float*)d_in[9];
    const float* bo   = (const float*)d_in[10];
    const float* qnw  = (const float*)d_in[11];
    const float* knw  = (const float*)d_in[12];
    const float* wout = (const float*)d_in[13];
    float* out = (float*)d_out;

    constexpr int M = 2 * 4096;      // 8192 token rows
    constexpr int K = 1024;

    char* p = (char*)d_ws;
    size_t off = 0;
    auto alloc = [&](size_t bytes) -> char* {
        char* r = p + off;
        off = (off + bytes + 255) & ~(size_t)255;
        return r;
    };

    __bf16* W1    = (__bf16*)alloc((size_t)2048 * 1024 * 2);
    __bf16* W2    = (__bf16*)alloc((size_t)6144 * 1024 * 2);
    __bf16* Wout  = (__bf16*)alloc((size_t)1024 * 1024 * 2);
    __bf16* xbf   = (__bf16*)alloc((size_t)M * 1024 * 2);
    float*  bias6 = (float*) alloc((size_t)6144 * 4);
    float*  xp    = (float*) alloc((size_t)M * 2048 * 4);
    __bf16* xin   = (__bf16*)alloc((size_t)M * 1024 * 2);
    float*  gsig  = (float*) alloc((size_t)M * 1024 * 4);
    float*  raw6  = (float*) alloc((size_t)M * 6144 * 4);
    float*  qb    = (float*) alloc((size_t)M * 1024 * 4);
    float*  kb    = (float*) alloc((size_t)M * 1024 * 4);
    float*  vb    = (float*) alloc((size_t)M * 1024 * 4);
    float*  ib2   = (float*) alloc((size_t)M * 1024 * 4);
    float*  fb2   = (float*) alloc((size_t)M * 1024 * 4);
    float*  ogrow = (float*) alloc((size_t)M * 1024 * 4);
    __bf16* hgate = (__bf16*)alloc((size_t)M * 1024 * 2);

    auto conv = [&](const float* src, __bf16* dst, int n) {
        f32_to_bf16<<<dim3((n + 255) / 256), dim3(256), 0, stream>>>(src, dst, n);
    };

    // 1) weight + activation conversion (fp32 -> bf16)
    conv(w_in, W1, 2048 * 1024);
    conv(wq, W2 + (size_t)0    * 1024 * 1024, 1024 * 1024);
    conv(wk, W2 + (size_t)1    * 1024 * 1024, 1024 * 1024);
    conv(wv, W2 + (size_t)2    * 1024 * 1024, 1024 * 1024);
    conv(wi, W2 + (size_t)3    * 1024 * 1024, 1024 * 1024);
    conv(wf, W2 + (size_t)4    * 1024 * 1024, 1024 * 1024);
    conv(wo, W2 + (size_t)5    * 1024 * 1024, 1024 * 1024);
    conv(wout, Wout, 1024 * 1024);
    conv(x, xbf, M * 1024);
    build_bias6<<<dim3(24), dim3(256), 0, stream>>>(bi, bfb, bo, bias6);

    // 2) in-projection GEMM: (8192 x 1024) x (2048 x 1024)^T
    gemm_bf16_wmma<<<dim3(2048 / TILE_N, M / TILE_M), dim3(256), 0, stream>>>(
        xbf, W1, nullptr, xp, M, 2048, K);

    // 3) split -> bf16 x_inner + sigmoid(x_gate)
    split_xp<<<dim3((M * 1024 + 255) / 256), dim3(256), 0, stream>>>(xp, xin, gsig, M * 1024);

    // 4) fused 6-way projection GEMM: (8192 x 1024) x (6144 x 1024)^T + bias
    gemm_bf16_wmma<<<dim3(6144 / TILE_N, M / TILE_M), dim3(256), 0, stream>>>(
        xin, W2, bias6, raw6, M, 6144, K);

    // 5) activations + per-head RMSNorm into scan layout
    activate_kernel<<<dim3(M), dim3(256), 0, stream>>>(
        raw6, qnw, knw, qb, kb, vb, ib2, fb2, ogrow);

    // 6) sequential mLSTM scan (32 blocks = B*H), fused output gating -> bf16
    mlstm_scan_kernel<<<dim3(32), dim3(256), 0, stream>>>(
        qb, kb, vb, ib2, fb2, ogrow, gsig, hgate);

    // 7) out-projection GEMM: (8192 x 1024) x (1024 x 1024)^T -> fp32 output
    gemm_bf16_wmma<<<dim3(1024 / TILE_N, M / TILE_M), dim3(256), 0, stream>>>(
        hgate, Wout, nullptr, out, M, 1024, K);
}
